// HyperConvNet_77549929497035
// MI455X (gfx1250) — compile-verified, matrix operand order checked by reading, not run on hardware
//
#include <hip/hip_runtime.h>
#include <hip/hip_bf16.h>
#include <math.h>

// ---------------- problem constants (from reference) ----------------
static constexpr int Bb      = 8192;
static constexpr int Tt      = 34;
static constexpr int MHC     = 64;
static constexpr int Hh      = 64;          // LSTM hidden
static constexpr int PEP     = 320;
static constexpr int NPER    = 24;
static constexpr int NN      = Bb * NPER;   // 196608 nodes
static constexpr int NHE     = 65536;
static constexpr int EE      = NN * 3;      // 589824 edges

// ---------------- WMMA types / helpers (gfx1250, wave32) ----------------
typedef __attribute__((ext_vector_type(16))) __bf16 bf16x16;
typedef __attribute__((ext_vector_type(8)))  float  floatx8;

__device__ __forceinline__ floatx8 wmma_bf16(bf16x16 a, bf16x16 b, floatx8 c) {
    // D = A(16x32) * B(32x16) + C, f32 accumulate
    return __builtin_amdgcn_wmma_f32_16x16x32_bf16(false, a, false, b, (short)0, c, false, false);
}

// Load a 16x32 tile (rows x K) of fp32 and convert to bf16 in the ISA A/B layout:
// lane&15 = row; lanes 0-15 hold K {0..7,16..23}, lanes 16-31 hold K {8..15,24..31}.
__device__ __forceinline__ bf16x16 ld_tile_f32(const float* __restrict__ base, int stride) {
    const int lane = threadIdx.x & 31;
    const float* r = base + (size_t)(lane & 15) * stride + ((lane & 16) ? 8 : 0);
    float4 f0 = *(const float4*)(r + 0);
    float4 f1 = *(const float4*)(r + 4);
    float4 f2 = *(const float4*)(r + 16);
    float4 f3 = *(const float4*)(r + 20);
    bf16x16 v;
    v[0]=(__bf16)f0.x; v[1]=(__bf16)f0.y; v[2]=(__bf16)f0.z; v[3]=(__bf16)f0.w;
    v[4]=(__bf16)f1.x; v[5]=(__bf16)f1.y; v[6]=(__bf16)f1.z; v[7]=(__bf16)f1.w;
    v[8]=(__bf16)f2.x; v[9]=(__bf16)f2.y; v[10]=(__bf16)f2.z; v[11]=(__bf16)f2.w;
    v[12]=(__bf16)f3.x; v[13]=(__bf16)f3.y; v[14]=(__bf16)f3.z; v[15]=(__bf16)f3.w;
    return v;
}

// Same layout from an LDS-resident bf16 array (row-major, given stride in elements).
__device__ __forceinline__ bf16x16 ld_tile_bf16s(const __bf16* __restrict__ base, int stride) {
    const int lane = threadIdx.x & 31;
    const __bf16* r = base + (size_t)(lane & 15) * stride + ((lane & 16) ? 8 : 0);
    bf16x16 v;
#pragma unroll
    for (int i = 0; i < 8; ++i) v[i]     = r[i];
#pragma unroll
    for (int i = 0; i < 8; ++i) v[8 + i] = r[16 + i];
    return v;
}

// Store 16x16 f32 accumulator: VGPR j -> row (j + 8*(lane>=16)), col = lane&15.
__device__ __forceinline__ void st_tile(float* __restrict__ base, int ldd, floatx8 c, int relu) {
    const int lane = threadIdx.x & 31;
    const int n  = lane & 15;
    const int mb = (lane & 16) ? 8 : 0;
#pragma unroll
    for (int j = 0; j < 8; ++j) {
        float v = c[j];
        if (relu) v = fmaxf(v, 0.f);
        base[(size_t)(mb + j) * ldd + n] = v;
    }
}

__device__ __forceinline__ float sigf(float x) { return 1.0f / (1.0f + __expf(-x)); }

// CDNA5 async global->LDS copy (16B per lane) + ASYNCcnt wait (inline asm: portable
// across ROCm-7.2 / amdgpu-toolchain builtin arity differences).
__device__ __forceinline__ void async_copy_b128(unsigned lds_off, const float* gptr) {
    asm volatile("global_load_async_to_lds_b128 %0, %1, off"
                 :: "v"(lds_off), "v"(gptr) : "memory");
}
__device__ __forceinline__ void wait_async0() {
    asm volatile("s_wait_asynccnt 0" ::: "memory");
}

// ---------------- generic GEMM: C[M,N] = A[M,K] * W[N,K]^T (+bias)(+relu) ----------------
// grid = (N/64, M/64), block = 256 (8 waves). A panel (64x32 f32 per k-step) is staged
// into LDS with GLOBAL_LOAD_ASYNC_TO_LDS_B128 and consumed by all 8 waves (1x A traffic).
// Wave (w&3) = 16-row M-subtile, (w>>2) = 32-col N group. W (small, L2-resident) direct.
__global__ void gemm_bf16_kernel(const float* __restrict__ A, const float* __restrict__ W,
                                 float* __restrict__ C, int M, int N, int K,
                                 const float* __restrict__ bias, int relu) {
    __shared__ float Apanel[64 * 32];                 // 8 KB

    const int wave = threadIdx.x >> 5;
    const int lane = threadIdx.x & 31;
    const int m0b  = blockIdx.y * 64;
    const int msub = (wave & 3) * 16;
    const int n0   = blockIdx.x * 64 + (wave >> 2) * 32;

    floatx8 c0, c1;
    float b0v = bias ? bias[n0 + (lane & 15)] : 0.f;
    float b1v = bias ? bias[n0 + 16 + (lane & 15)] : 0.f;
#pragma unroll
    for (int j = 0; j < 8; ++j) { c0[j] = b0v; c1[j] = b1v; }

    const float* W0 = W + (size_t)n0 * K;
    const float* W1 = W + (size_t)(n0 + 16) * K;

    // per-thread staging slots: 512 x b128 per panel, 2 per thread
    const int r0 = threadIdx.x >> 3,        s0 = (threadIdx.x & 7) * 4;
    const int r1 = (threadIdx.x + 256) >> 3, s1 = s0;   // idx+256: row += 32, same seg
    const unsigned ldsA = (unsigned)(size_t)(const void*)Apanel;

    for (int k0 = 0; k0 < K; k0 += 32) {
        // async-stage A[m0b..m0b+63][k0..k0+31] -> LDS
        async_copy_b128(ldsA + (unsigned)(r0 * 32 + s0) * 4,
                        A + (size_t)(m0b + r0) * K + k0 + s0);
        async_copy_b128(ldsA + (unsigned)(r1 * 32 + s1) * 4,
                        A + (size_t)(m0b + r1) * K + k0 + s1);
        wait_async0();
        __syncthreads();

        bf16x16 a  = ld_tile_f32(Apanel + msub * 32, 32);
        bf16x16 w0 = ld_tile_f32(W0 + k0, K);
        bf16x16 w1 = ld_tile_f32(W1 + k0, K);
        c0 = wmma_bf16(a, w0, c0);
        c1 = wmma_bf16(a, w1, c1);
        __syncthreads();
    }
    st_tile(C + (size_t)(m0b + msub) * N + n0,      N, c0, relu);
    st_tile(C + (size_t)(m0b + msub) * N + n0 + 16, N, c1, relu);
}

// ---------------- fused BiLSTM direction kernel ----------------
// One block = 16 batch rows, 256 threads = 8 waves (each wave: 32 of the 256 gate cols).
// Combined weights [w_ih | w_hh] converted once to bf16 in LDS; recurrence stays resident.
__global__ void lstm_dir_kernel(const float* __restrict__ x, int in_dim,
                                const float* __restrict__ w_ih, const float* __restrict__ w_hh,
                                const float* __restrict__ b_ih, const float* __restrict__ b_hh,
                                float* __restrict__ hout, int reverse, int dirOff) {
    extern __shared__ unsigned char smem[];
    const int KC = in_dim + Hh;                       // 128 or 192, multiple of 32
    __bf16* wcomb = (__bf16*)smem;                    // 256 * KC bf16
    float*  biasS = (float*)(smem + (size_t)256 * KC * sizeof(__bf16)); // 256
    float*  hbuf  = biasS + 256;                      // 16*64
    float*  cbuf  = hbuf + 16 * Hh;                   // 16*64
    float*  gates = cbuf + 16 * Hh;                   // 16*256

    const int tid = threadIdx.x;
    const int b0  = blockIdx.x * 16;

    for (int i = tid; i < 256 * KC; i += 256) {
        int col = i / KC, k = i % KC;
        float w = (k < in_dim) ? w_ih[(size_t)col * in_dim + k]
                               : w_hh[(size_t)col * Hh + (k - in_dim)];
        wcomb[i] = (__bf16)w;
    }
    biasS[tid] = b_ih[tid] + b_hh[tid];
    for (int i = tid; i < 16 * Hh; i += 256) { hbuf[i] = 0.f; cbuf[i] = 0.f; }
    __syncthreads();

    const int wave = tid >> 5, lane = tid & 31;
    const int n0 = wave * 32;
    const int xstride = Tt * in_dim;
    const float bv0 = biasS[n0 + (lane & 15)];
    const float bv1 = biasS[n0 + 16 + (lane & 15)];

    for (int ts = 0; ts < Tt; ++ts) {
        const int t = reverse ? (Tt - 1 - ts) : ts;
        floatx8 c0, c1;
#pragma unroll
        for (int j = 0; j < 8; ++j) { c0[j] = bv0; c1[j] = bv1; }

        const float* xb = x + ((size_t)b0 * Tt + t) * in_dim;
        for (int k0 = 0; k0 < KC; k0 += 32) {
            bf16x16 a = (k0 < in_dim) ? ld_tile_f32(xb + k0, xstride)
                                      : ld_tile_f32(hbuf + (k0 - in_dim), Hh);
            bf16x16 wb0 = ld_tile_bf16s(wcomb + (size_t)n0 * KC + k0, KC);
            bf16x16 wb1 = ld_tile_bf16s(wcomb + (size_t)(n0 + 16) * KC + k0, KC);
            c0 = wmma_bf16(a, wb0, c0);
            c1 = wmma_bf16(a, wb1, c1);
        }
        st_tile(gates + n0,      256, c0, 0);
        st_tile(gates + n0 + 16, 256, c1, 0);
        __syncthreads();

#pragma unroll
        for (int q = 0; q < 4; ++q) {
            int idx = tid + q * 256;          // 0..1023 -> (row, hidden j)
            int row = idx >> 6, j = idx & 63;
            float ig = sigf(gates[row * 256 + j]);
            float fg = sigf(gates[row * 256 + 64 + j]);
            float gg = tanhf(gates[row * 256 + 128 + j]);
            float og = sigf(gates[row * 256 + 192 + j]);
            float cc = fg * cbuf[row * Hh + j] + ig * gg;
            float hh = og * tanhf(cc);
            cbuf[row * Hh + j] = cc;
            hbuf[row * Hh + j] = hh;
            hout[((size_t)(b0 + row) * Tt + t) * (2 * Hh) + dirOff + j] = hh;
        }
        __syncthreads();
    }
}

// ---------------- hypergraph helpers ----------------
__global__ void degree_kernel(const int* __restrict__ row, const int* __restrict__ col,
                              float* __restrict__ degv, float* __restrict__ dege, int E) {
    int e = blockIdx.x * blockDim.x + threadIdx.x;
    if (e < E) {
        atomicAdd(&degv[row[e]], 1.0f);
        atomicAdd(&dege[col[e]], 1.0f);
    }
}

// one 320-thread block per edge; coalesced gather + atomic scatter of a 320-vector
__global__ void scatter320_kernel(const float* __restrict__ src, const int* __restrict__ sidx,
                                  const int* __restrict__ didx, float* __restrict__ dst) {
    int e = blockIdx.x;
    int s = sidx[e], d = didx[e];
    atomicAdd(&dst[(size_t)d * PEP + threadIdx.x], src[(size_t)s * PEP + threadIdx.x]);
}

__global__ void scale_rows_kernel(float* __restrict__ buf, const float* __restrict__ deg) {
    int r = blockIdx.x;
    float dv = deg[r];
    float inv = dv > 0.f ? 1.f / dv : 0.f;
    buf[(size_t)r * PEP + threadIdx.x] *= inv;
}

__global__ void finalize_rows_kernel(float* __restrict__ buf, const float* __restrict__ deg,
                                     const float* __restrict__ bias, int relu) {
    int r = blockIdx.x;
    float dv = deg[r];
    float inv = dv > 0.f ? 1.f / dv : 0.f;
    float v = buf[(size_t)r * PEP + threadIdx.x] * inv + bias[threadIdx.x];
    if (relu) v = fmaxf(v, 0.f);
    buf[(size_t)r * PEP + threadIdx.x] = v;
}

// ---------------- clus_map: per-batch (34x128)·(24x128)^T ----------------
__global__ void clus_kernel(const float* __restrict__ mhc, const float* __restrict__ pep,
                            float* __restrict__ clus) {
    int b = blockIdx.x;
    for (int idx = threadIdx.x; idx < Tt * NPER; idx += blockDim.x) {
        int t = idx / NPER, n = idx % NPER;
        const float* a = mhc + ((size_t)b * Tt + t) * 128;
        const float* p = pep + ((size_t)b * NPER + n) * 128;
        float s = 0.f;
#pragma unroll 4
        for (int k = 0; k < 128; ++k) s = fmaf(a[k], p[k], s);
        clus[((size_t)b * Tt + t) * NPER + n] = s;
    }
}

// ---------------- head: fc1(816->16)+relu then out(16->1) ----------------
__global__ void head_kernel(const float* __restrict__ clus, const float* __restrict__ fc1_w,
                            const float* __restrict__ fc1_b, const float* __restrict__ out_w,
                            const float* __restrict__ out_b, float* __restrict__ out) {
    int b = blockIdx.x;
    __shared__ float partial[64];
    __shared__ float xc[16];
    int j = threadIdx.x >> 2, p = threadIdx.x & 3;
    float s = 0.f;
    const float* cb = clus + (size_t)b * (Tt * NPER);
    for (int i = p; i < Tt * NPER; i += 4) s = fmaf(cb[i], fc1_w[(size_t)j * (Tt * NPER) + i], s);
    partial[threadIdx.x] = s;
    __syncthreads();
    if (p == 0) {
        float v = partial[threadIdx.x] + partial[threadIdx.x + 1] +
                  partial[threadIdx.x + 2] + partial[threadIdx.x + 3] + fc1_b[j];
        xc[j] = fmaxf(v, 0.f);
    }
    __syncthreads();
    if (threadIdx.x == 0) {
        float s2 = out_b[0];
#pragma unroll
        for (int k = 0; k < 16; ++k) s2 = fmaf(xc[k], out_w[k], s2);
        out[b] = s2;
    }
}

// ---------------- launch ----------------
extern "C" void kernel_launch(void* const* d_in, const int* in_sizes, int n_in,
                              void* d_out, int out_size, void* d_ws, size_t ws_size,
                              hipStream_t stream) {
    (void)in_sizes; (void)n_in; (void)out_size; (void)ws_size;
    const float* x        = (const float*)d_in[0];
    const int*   edge     = (const int*)d_in[1];
    const int*   row      = edge;
    const int*   col      = edge + EE;
    const float* mhc      = (const float*)d_in[2];
    const float* wih0f = (const float*)d_in[3],  *whh0f = (const float*)d_in[4];
    const float* bih0f = (const float*)d_in[5],  *bhh0f = (const float*)d_in[6];
    const float* wih0b = (const float*)d_in[7],  *whh0b = (const float*)d_in[8];
    const float* bih0b = (const float*)d_in[9],  *bhh0b = (const float*)d_in[10];
    const float* wih1f = (const float*)d_in[11], *whh1f = (const float*)d_in[12];
    const float* bih1f = (const float*)d_in[13], *bhh1f = (const float*)d_in[14];
    const float* wih1b = (const float*)d_in[15], *whh1b = (const float*)d_in[16];
    const float* bih1b = (const float*)d_in[17], *bhh1b = (const float*)d_in[18];
    const float* hc1_w = (const float*)d_in[19], *hc1_b = (const float*)d_in[20];
    const float* hc2_w = (const float*)d_in[21], *hc2_b = (const float*)d_in[22];
    const float* pfw   = (const float*)d_in[23], *pfb   = (const float*)d_in[24];
    const float* fc1_w = (const float*)d_in[25], *fc1_b = (const float*)d_in[26];
    const float* out_w = (const float*)d_in[27], *out_b = (const float*)d_in[28];

    float* outp = (float*)d_out;          // [B]
    float* clus = outp + Bb;              // [B,34,24]

    // workspace carve (floats): h0,h1: 35.7M each; bufA,bufB: 62.9M each; efeat 21M; degs
    float* ws = (float*)d_ws;
    size_t o = 0;
    float* h0   = ws + o; o += (size_t)Bb * Tt * 128;
    float* h1   = ws + o; o += (size_t)Bb * Tt * 128;
    float* bufA = ws + o; o += (size_t)NN * PEP;
    float* bufB = ws + o; o += (size_t)NN * PEP;
    float* ef   = ws + o; o += (size_t)NHE * PEP;
    float* degv = ws + o; o += (size_t)NN;
    float* dege = ws + o; o += (size_t)NHE;

    auto smem_for = [](int in_dim) {
        int KC = in_dim + Hh;
        return (size_t)256 * KC * sizeof(__bf16) + 256 * 4 + 2 * 16 * Hh * 4 + 16 * 256 * 4;
    };
    const size_t sh0 = smem_for(MHC);       // ~91 KB
    const size_t sh1 = smem_for(2 * Hh);    // ~124 KB (fits 320 KB WGP LDS)

    // ---- BiLSTM (layer 0 then layer 1), each direction fused & LDS-resident ----
    lstm_dir_kernel<<<Bb / 16, 256, sh0, stream>>>(mhc, MHC, wih0f, whh0f, bih0f, bhh0f, h0, 0, 0);
    lstm_dir_kernel<<<Bb / 16, 256, sh0, stream>>>(mhc, MHC, wih0b, whh0b, bih0b, bhh0b, h0, 1, Hh);
    lstm_dir_kernel<<<Bb / 16, 256, sh1, stream>>>(h0, 2 * Hh, wih1f, whh1f, bih1f, bhh1f, h1, 0, 0);
    lstm_dir_kernel<<<Bb / 16, 256, sh1, stream>>>(h0, 2 * Hh, wih1b, whh1b, bih1b, bhh1b, h1, 1, Hh);

    // ---- degrees ----
    hipMemsetAsync(degv, 0, (size_t)(NN + NHE) * sizeof(float), stream);
    degree_kernel<<<(EE + 255) / 256, 256, 0, stream>>>(row, col, degv, dege, EE);

    // ---- hyperconv layer 1: xt = x @ hc1_w^T ; V->E ; E->V ; *d_inv + b ; relu ----
    gemm_bf16_kernel<<<dim3(PEP / 64, NN / 64), 256, 0, stream>>>(x, hc1_w, bufA, NN, PEP, PEP, nullptr, 0);
    hipMemsetAsync(ef, 0, (size_t)NHE * PEP * sizeof(float), stream);
    scatter320_kernel<<<EE, PEP, 0, stream>>>(bufA, row, col, ef);
    scale_rows_kernel<<<NHE, PEP, 0, stream>>>(ef, dege);
    hipMemsetAsync(bufB, 0, (size_t)NN * PEP * sizeof(float), stream);
    scatter320_kernel<<<EE, PEP, 0, stream>>>(ef, col, row, bufB);
    finalize_rows_kernel<<<NN, PEP, 0, stream>>>(bufB, degv, hc1_b, 1);

    // ---- hyperconv layer 2 (no relu) ----
    gemm_bf16_kernel<<<dim3(PEP / 64, NN / 64), 256, 0, stream>>>(bufB, hc2_w, bufA, NN, PEP, PEP, nullptr, 0);
    hipMemsetAsync(ef, 0, (size_t)NHE * PEP * sizeof(float), stream);
    scatter320_kernel<<<EE, PEP, 0, stream>>>(bufA, row, col, ef);
    scale_rows_kernel<<<NHE, PEP, 0, stream>>>(ef, dege);
    hipMemsetAsync(bufB, 0, (size_t)NN * PEP * sizeof(float), stream);
    scatter320_kernel<<<EE, PEP, 0, stream>>>(ef, col, row, bufB);
    finalize_rows_kernel<<<NN, PEP, 0, stream>>>(bufB, degv, hc2_b, 0);

    // ---- pep = xg @ pep_fc_w^T + b  -> bufA [NN,128] ----
    gemm_bf16_kernel<<<dim3(128 / 64, NN / 64), 256, 0, stream>>>(bufB, pfw, bufA, NN, 128, PEP, pfb, 0);

    // ---- clus_map + head ----
    clus_kernel<<<Bb, 256, 0, stream>>>(h1, bufA, clus);
    head_kernel<<<Bb, 64, 0, stream>>>(clus, fc1_w, fc1_b, out_w, out_b, outp);
}
